// TopKSAE_53618371723774
// MI455X (gfx1250) — matrix operand order repaired and checked by reading, not compiled, not run on hardware
//
#include <hip/hip_runtime.h>
#include <hip/hip_bf16.h>
#include <stdint.h>

#define D_IN   768
#define D_DICT 16384
#define KTOP   32
#define NROWS  8192

typedef __attribute__((ext_vector_type(16))) __bf16 v16bf;
typedef __attribute__((ext_vector_type(8)))  __bf16 v8bf;
typedef __attribute__((ext_vector_type(8)))  float  v8f;

// ---------------------------------------------------------------------------
// fp32 -> bf16 (RNE) split into hi + lo parts, done with integer bit ops so we
// never rely on scalar __bf16 arithmetic support.
// ---------------------------------------------------------------------------
__device__ __forceinline__ void split_bf16(float f, unsigned short* h, unsigned short* l) {
    unsigned u  = __float_as_uint(f);
    unsigned hr = (u + 0x7FFFu + ((u >> 16) & 1u)) & 0xFFFF0000u;   // RNE to bf16
    float fh    = __uint_as_float(hr);
    float fl    = f - fh;                                           // exact residual
    unsigned ul = __float_as_uint(fl);
    unsigned lr = (ul + 0x7FFFu + ((ul >> 16) & 1u));
    *h = (unsigned short)(hr >> 16);
    *l = (unsigned short)(lr >> 16);
}

__device__ __forceinline__ unsigned fkey(float f) {
    unsigned b = __float_as_uint(f);
    return b ^ ((b & 0x80000000u) ? 0xFFFFFFFFu : 0x80000000u);     // order-preserving
}

#define SHUF16(a, b) __builtin_shufflevector((a), (b), 0,1,2,3,4,5,6,7,8,9,10,11,12,13,14,15)

// ---------------------------------------------------------------------------
// Kernel 1: z[n, m] = sum_d x[n,d] * W_enc[m,d] + b_enc[m]
// 128x128 block tile, BK=32, split-bf16 WMMA (3 wmma per 16x16 tile per K step)
// ---------------------------------------------------------------------------
__global__ __launch_bounds__(256)
void sae_encode_gemm(const float* __restrict__ x,
                     const float* __restrict__ Wenc,
                     const float* __restrict__ benc,
                     float* __restrict__ z)
{
    __shared__ __attribute__((aligned(16))) unsigned short Ah[128 * 32]; // [m][k]
    __shared__ __attribute__((aligned(16))) unsigned short Al[128 * 32];
    __shared__ __attribute__((aligned(16))) unsigned short Bh[32 * 128]; // [n][k]
    __shared__ __attribute__((aligned(16))) unsigned short Bl[32 * 128];

    const int tid     = threadIdx.x;
    const int rowbase = blockIdx.y * 128;   // batch rows
    const int colbase = blockIdx.x * 128;   // dict cols
    const int wid     = tid >> 5;
    const int lane    = tid & 31;
    const int wr      = wid >> 2;           // 0..1 : 64-row slab
    const int wc      = wid & 3;            // 0..3 : 32-col slab

    v8f zero = {};
    v8f acc[4][2];
    #pragma unroll
    for (int i = 0; i < 4; ++i)
        #pragma unroll
        for (int j = 0; j < 2; ++j) acc[i][j] = zero;

    const int mr = lane & 15;
    const int kh = (lane >> 4) * 8;         // A-frag K sub-offset per lane half

    for (int kb = 0; kb < D_IN / 32; ++kb) {
        // ---- stage A tile (x rows, K-contiguous) as bf16 hi/lo ----
        #pragma unroll
        for (int it = 0; it < 16; ++it) {
            int idx = tid + it * 256;
            int r = idx >> 5, c = idx & 31;
            const float* gp = &x[(size_t)(rowbase + r) * D_IN + kb * 32 + c];
            if (kb + 1 < D_IN / 32) __builtin_prefetch(gp + 32);
            unsigned short h, l;
            split_bf16(*gp, &h, &l);
            Ah[r * 32 + c] = h;  Al[r * 32 + c] = l;
        }
        // ---- stage B tile (W_enc rows) as [n][k] bf16 hi/lo ----
        #pragma unroll
        for (int it = 0; it < 16; ++it) {
            int idx = tid + it * 256;
            int m = idx >> 5, d = idx & 31;
            const float* gp = &Wenc[(size_t)(colbase + m) * D_IN + kb * 32 + d];
            if (kb + 1 < D_IN / 32) __builtin_prefetch(gp + 32);
            unsigned short h, l;
            split_bf16(*gp, &h, &l);
            Bh[m * 32 + d] = h;  Bl[m * 32 + d] = l;
        }
        __syncthreads();

        // ---- A fragments: lanes 0-15 hold K={kh..kh+7, 16+kh..}, per ISA layout ----
        v16bf afh[4], afl[4];
        #pragma unroll
        for (int tm = 0; tm < 4; ++tm) {
            int m = wr * 64 + tm * 16 + mr;
            afh[tm] = SHUF16(*(const v8bf*)&Ah[m * 32 + kh],
                             *(const v8bf*)&Ah[m * 32 + 16 + kh]);
            afl[tm] = SHUF16(*(const v8bf*)&Al[m * 32 + kh],
                             *(const v8bf*)&Al[m * 32 + 16 + kh]);
        }
        // ---- B fragments: lane holds column N=lane&15, K={kh..kh+7, 16+kh..} ----
        #pragma unroll
        for (int tn = 0; tn < 2; ++tn) {
            int n0 = wc * 32 + tn * 16;
            v16bf bfh = SHUF16(*(const v8bf*)&Bh[(n0 + mr) * 32 + kh],
                               *(const v8bf*)&Bh[(n0 + mr) * 32 + 16 + kh]);
            v16bf bfl = SHUF16(*(const v8bf*)&Bl[(n0 + mr) * 32 + kh],
                               *(const v8bf*)&Bl[(n0 + mr) * 32 + 16 + kh]);
            #pragma unroll
            for (int tm = 0; tm < 4; ++tm) {
                acc[tm][tn] = __builtin_amdgcn_wmma_f32_16x16x32_bf16(
                    false, afh[tm], false, bfh, (short)0, acc[tm][tn], false, false);
                acc[tm][tn] = __builtin_amdgcn_wmma_f32_16x16x32_bf16(
                    false, afh[tm], false, bfl, (short)0, acc[tm][tn], false, false);
                acc[tm][tn] = __builtin_amdgcn_wmma_f32_16x16x32_bf16(
                    false, afl[tm], false, bfh, (short)0, acc[tm][tn], false, false);
            }
        }
        __syncthreads();
    }

    // ---- epilogue: bias + store (C/D layout: M = r + 8*(lane>>4), N = lane&15) ----
    #pragma unroll
    for (int tn = 0; tn < 2; ++tn) {
        int col   = colbase + wc * 32 + tn * 16 + (lane & 15);
        float bias = benc[col];
        #pragma unroll
        for (int tm = 0; tm < 4; ++tm) {
            int mbase = rowbase + wr * 64 + tm * 16 + (lane >> 4) * 8;
            #pragma unroll
            for (int r = 0; r < 8; ++r)
                z[(size_t)(mbase + r) * D_DICT + col] = acc[tm][tn][r] + bias;
        }
    }
}

// ---------------------------------------------------------------------------
// Kernel 2: exact top-K per row via bitwise radix-select on order-preserving
// keys, then deterministic compaction of the K winners + in-place masking.
// One 256-thread block per row; 64 values per thread held in registers.
// ---------------------------------------------------------------------------
__global__ __launch_bounds__(256)
void sae_topk(float* __restrict__ z,
              float* __restrict__ wsVal,
              int*   __restrict__ wsIdx)
{
    __shared__ unsigned cnt;
    __shared__ unsigned scan[256];

    const int row = blockIdx.x;
    const int tid = threadIdx.x;
    const size_t base = (size_t)row * D_DICT;

    float v[64];
    #pragma unroll
    for (int j = 0; j < 64; ++j) v[j] = z[base + (size_t)j * 256 + tid];

    // T = key of the K-th largest element (exact)
    unsigned T = 0;
    for (int bit = 31; bit >= 0; --bit) {
        unsigned cand = T | (1u << bit);
        unsigned local = 0;
        #pragma unroll
        for (int j = 0; j < 64; ++j) local += (fkey(v[j]) >= cand) ? 1u : 0u;
        if (tid == 0) cnt = 0;
        __syncthreads();
        atomicAdd(&cnt, local);
        __syncthreads();
        unsigned total = cnt;
        __syncthreads();
        if (total >= KTOP) T = cand;
    }

    // deterministic exclusive scan of per-thread keep counts
    unsigned myc = 0;
    #pragma unroll
    for (int j = 0; j < 64; ++j) myc += (fkey(v[j]) >= T) ? 1u : 0u;
    scan[tid] = myc;
    __syncthreads();
    for (int off = 1; off < 256; off <<= 1) {
        unsigned t = (tid >= off) ? scan[tid - off] : 0u;
        __syncthreads();
        scan[tid] += t;
        __syncthreads();
    }
    unsigned pos = scan[tid] - myc;   // exclusive prefix

    #pragma unroll
    for (int j = 0; j < 64; ++j) {
        int gidx = j * 256 + tid;
        if (fkey(v[j]) >= T) {
            if (pos < KTOP) {                      // keep (z already holds v)
                wsVal[row * KTOP + pos] = v[j];
                wsIdx[row * KTOP + pos] = gidx;
            } else {                               // tie overflow beyond K
                z[base + gidx] = 0.0f;
            }
            pos++;
        } else {
            z[base + gidx] = 0.0f;                 // masked
        }
    }
}

// ---------------------------------------------------------------------------
// Kernel 3: x_hat[n,d] = b_dec[d] + sum_k val_k * W_dec[d, idx_k]
// One block per row; thread t owns d = t, t+256, t+512.
// ---------------------------------------------------------------------------
__global__ __launch_bounds__(256)
void sae_decode(const float* __restrict__ Wdec,
                const float* __restrict__ bdec,
                const float* __restrict__ wsVal,
                const int*   __restrict__ wsIdx,
                float* __restrict__ xhat)
{
    __shared__ float lv[KTOP];
    __shared__ int   li[KTOP];
    const int row = blockIdx.x;
    const int tid = threadIdx.x;
    if (tid < KTOP) { lv[tid] = wsVal[row * KTOP + tid]; li[tid] = wsIdx[row * KTOP + tid]; }
    __syncthreads();

    float a0 = bdec[tid];
    float a1 = bdec[tid + 256];
    float a2 = bdec[tid + 512];
    #pragma unroll
    for (int k = 0; k < KTOP; ++k) {
        float s = lv[k]; int id = li[k];
        a0 = fmaf(s, Wdec[(size_t)(tid      ) * D_DICT + id], a0);
        a1 = fmaf(s, Wdec[(size_t)(tid + 256) * D_DICT + id], a1);
        a2 = fmaf(s, Wdec[(size_t)(tid + 512) * D_DICT + id], a2);
    }
    xhat[(size_t)row * D_IN + tid      ] = a0;
    xhat[(size_t)row * D_IN + tid + 256] = a1;
    xhat[(size_t)row * D_IN + tid + 512] = a2;
}

// ---------------------------------------------------------------------------
extern "C" void kernel_launch(void* const* d_in, const int* in_sizes, int n_in,
                              void* d_out, int out_size, void* d_ws, size_t ws_size,
                              hipStream_t stream)
{
    (void)in_sizes; (void)n_in; (void)out_size; (void)ws_size;
    const float* x    = (const float*)d_in[0];
    const float* Wenc = (const float*)d_in[1];
    const float* benc = (const float*)d_in[2];
    const float* Wdec = (const float*)d_in[3];
    const float* bdec = (const float*)d_in[4];

    float* xhat = (float*)d_out;                               // [N, D_IN]
    float* z    = (float*)d_out + (size_t)NROWS * D_IN;        // [N, D_DICT]

    float* wsVal = (float*)d_ws;                               // [N, K]
    int*   wsIdx = (int*)((char*)d_ws + (size_t)NROWS * KTOP * sizeof(float));

    dim3 g1(D_DICT / 128, NROWS / 128);                        // 128 x 64 blocks
    sae_encode_gemm<<<g1, 256, 0, stream>>>(x, Wenc, benc, z);
    sae_topk      <<<NROWS, 256, 0, stream>>>(z, wsVal, wsIdx);
    sae_decode    <<<NROWS, 256, 0, stream>>>(Wdec, bdec, wsVal, wsIdx, xhat);
}
